// MyModel_32762010533888
// MI455X (gfx1250) — matrix operand order repaired and checked by reading, not compiled, not run on hardware
//
#include <hip/hip_runtime.h>
#include <hip/hip_bf16.h>
#include <math.h>

#define BATCH 64
#define NN    512
#define LITW  128
#define SEMW  64
#define ED    64
#define HD    32
#define GATE  128   // 4*HD
#define OD    64
#define ITERS 5

typedef __attribute__((ext_vector_type(16))) _Float16 v16h;
typedef __attribute__((ext_vector_type(8)))  float    v8f;

// ---------------------------------------------------------------- WMMA core
__device__ __forceinline__ v8f wmma16(v16h a, v16h b, v8f c) {
  // D(16x16,f32) = A(16x32,f16) * B(32x16,f16) + C
  return __builtin_amdgcn_wmma_f32_16x16x32_f16(
      false, a, false, b, (short)0, c, false, false);
}

// Fragment tiles are stored pre-swizzled: 512 halfs per 16x32 (A) or 32x16 (B)
// tile, lane-major: lane's 16 halfs are contiguous -> one 32B load per lane.
__device__ __forceinline__ v16h load_frag(const _Float16* tile) {
  return *(const v16h*)(tile + ((threadIdx.x & 31) << 4));
}

// A fragment (16x32 MxK) gathered from a row-major f32 source (setup-only paths).
// frag[lane][e] = A[lane&15][(e&7) + ((e>>3)<<4) + ((lane>>4)<<3)]
__device__ __forceinline__ v16h load_a_f32(const float* A, int lda) {
  int lane = threadIdx.x & 31;
  const float* p = A + (lane & 15) * lda + ((lane >> 4) << 3);
  v16h a;
#pragma unroll
  for (int e = 0; e < 16; ++e) a[e] = (_Float16)p[(e & 7) + ((e >> 3) << 4)];
  return a;
}

// C/D fragment (16x16 f32): vgpr v -> row v + 8*(lane>=16), col = lane&15.
__device__ __forceinline__ v8f load_c_f32(const float* C, int ldc) {
  int lane = threadIdx.x & 31;
  const float* p = C + ((lane >> 4) << 3) * ldc + (lane & 15);
  v8f c;
#pragma unroll
  for (int v = 0; v < 8; ++v) c[v] = p[v * ldc];
  return c;
}
__device__ __forceinline__ void store_d_f32(float* D, int ldd, v8f d) {
  int lane = threadIdx.x & 31;
  float* p = D + ((lane >> 4) << 3) * ldd + (lane & 15);
#pragma unroll
  for (int v = 0; v < 8; ++v) p[v * ldd] = d[v];
}

// Store a D fragment into an A-fragment-swizzled f16 region covering a
// 16-row x 64-K block (two 16x32 tiles).  kcol = feature base of this D tile.
__device__ __forceinline__ void store_d_aswz(_Float16* base, int kcol, v8f d) {
  int lane = threadIdx.x & 31;
  int c  = lane & 15;
  int rb = (lane >> 4) << 3;
  int k  = kcol + c;
  int ks = k >> 5, kk = k & 31;
  int e  = (kk & 7) + ((kk >> 4) << 3);
  _Float16* p = base + ks * 512 + (rb + ((kk >> 3) & 1) * 16) * 16 + e;
#pragma unroll
  for (int v = 0; v < 8; ++v) p[v * 16] = (_Float16)d[v];
}

__device__ __forceinline__ float sigm(float x) { return 1.0f / (1.0f + __expf(-x)); }

// ---------------------------------------------------------------- setup swizzles
// B-fragment swizzle of a row-major f32 [K,N] weight -> f16 tiles (kt*(N/16)+nt).
__global__ void k_swz_b(const float* __restrict__ src, _Float16* __restrict__ dst,
                        int K, int N) {
  int o = blockIdx.x * blockDim.x + threadIdx.x;
  if (o >= K * N) return;
  int tile = o >> 9, lane = (o >> 4) & 31, e = o & 15;
  int tn = N >> 4;
  int kt = tile / tn, nt = tile - kt * tn;
  int k = kt * 32 + ((lane >> 4) << 3) + (e & 7) + ((e >> 3) << 4);  // B: table K order
  int c = nt * 16 + (lane & 15);
  dst[o] = (_Float16)src[k * N + c];
}

// A-fragment swizzle of the int 0/1 adjacency -> f16 (lossless).
// tiles: ((b*32 + rt)*16 + ks), rt = 16-row tile, ks = 32-K block.
__global__ void k_swz_adj(const int* __restrict__ cfg, _Float16* __restrict__ dst) {
  size_t o = (size_t)blockIdx.x * 256 + threadIdx.x;     // B*NN*NN elements
  int tile = (int)(o >> 9), lane = (int)(o >> 4) & 31, e = (int)o & 15;
  int b = tile >> 9;                                     // 512 tiles per batch
  int rem = tile & 511;
  int rt = rem >> 4, ks = rem & 15;
  int row = rt * 16 + (lane & 15);
  int k = ks * 32 + (e & 7) + ((e >> 3) << 4) + ((lane >> 4) << 3);
  dst[o] = (_Float16)(float)cfg[((size_t)b * NN + row) * NN + k];
}

// x0 = relu(bb) written straight into B-fragment-swizzled f16 layout.
// tiles: ((b*16 + kb)*4 + nt), kb = 32-node K block, nt = feature tile.
__global__ void k_init_x(const float* __restrict__ bb, _Float16* __restrict__ xswz) {
  int o = blockIdx.x * 256 + threadIdx.x;                // B*NN*ED elements
  int tile = o >> 9, lane = (o >> 4) & 31, e = o & 15;
  int nt = tile & 3, kb = (tile >> 2) & 15, b = tile >> 6;
  int n = kb * 32 + ((lane >> 4) << 3) + (e & 7) + ((e >> 3) << 4);
  int c = nt * 16 + (lane & 15);
  xswz[o] = (_Float16)fmaxf(bb[((size_t)b * NN + n) * ED + c], 0.0f);
}

// ---------------------------------------------------------------- branch MLP
// relu(relu(lit@d1w+b)@d2w+b) -> bb[:, :, 0:HD].  One wave = 16 rows.
__global__ void k_mlp(const float* __restrict__ lit,
                      const _Float16* __restrict__ w1s, const float* __restrict__ d1b,
                      const _Float16* __restrict__ w2s, const float* __restrict__ d2b,
                      float* __restrict__ bb) {
  __shared__ __align__(32) _Float16 u[2 * 512];  // 16 rows x K64, A-swizzled
  const int R   = blockIdx.x * 16;
  const int col = threadIdx.x & 15;
  v16h a0 = load_a_f32(lit + (size_t)R * LITW + 0,  LITW);
  v16h a1 = load_a_f32(lit + (size_t)R * LITW + 32, LITW);
  v16h a2 = load_a_f32(lit + (size_t)R * LITW + 64, LITW);
  v16h a3 = load_a_f32(lit + (size_t)R * LITW + 96, LITW);
#pragma unroll
  for (int nt = 0; nt < 4; ++nt) {
    v8f acc = {};
    acc = wmma16(a0, load_frag(w1s + (0 * 4 + nt) * 512), acc);
    acc = wmma16(a1, load_frag(w1s + (1 * 4 + nt) * 512), acc);
    acc = wmma16(a2, load_frag(w1s + (2 * 4 + nt) * 512), acc);
    acc = wmma16(a3, load_frag(w1s + (3 * 4 + nt) * 512), acc);
    float bias = d1b[nt * 16 + col];
#pragma unroll
    for (int v = 0; v < 8; ++v) acc[v] = fmaxf(acc[v] + bias, 0.0f);
    store_d_aswz(u, nt * 16, acc);
  }
  __syncthreads();
  v16h ua0 = load_frag(u), ua1 = load_frag(u + 512);
#pragma unroll
  for (int nt = 0; nt < 2; ++nt) {
    v8f acc = {};
    acc = wmma16(ua0, load_frag(w2s + (0 * 2 + nt) * 512), acc);
    acc = wmma16(ua1, load_frag(w2s + (1 * 2 + nt) * 512), acc);
    float bias = d2b[nt * 16 + col];
#pragma unroll
    for (int v = 0; v < 8; ++v) acc[v] = fmaxf(acc[v] + bias, 0.0f);
    store_d_f32(bb + (size_t)R * ED + nt * 16, ED, acc);
  }
}

// ---------------------------------------------------------------- LSTM input proj
// xkT[t, batch, gate] = sem @ lk + lb (time-major for the recurrence).
__global__ void k_xk(const float* __restrict__ sem, const _Float16* __restrict__ lks,
                     const float* __restrict__ lb, float* __restrict__ xkT) {
  const int R    = blockIdx.x * 16;
  const int lane = threadIdx.x & 31;
  const int col  = lane & 15;
  const int rb   = (lane >> 4) << 3;
  v16h a0 = load_a_f32(sem + (size_t)R * SEMW + 0,  SEMW);
  v16h a1 = load_a_f32(sem + (size_t)R * SEMW + 32, SEMW);
#pragma unroll
  for (int nt = 0; nt < 8; ++nt) {
    v8f acc = {};
    acc = wmma16(a0, load_frag(lks + (0 * 8 + nt) * 512), acc);
    acc = wmma16(a1, load_frag(lks + (1 * 8 + nt) * 512), acc);
    float bias = lb[nt * 16 + col];
#pragma unroll
    for (int v = 0; v < 8; ++v) {
      int gr = R + rb + v;                 // flattened b*NN + n
      int b = gr >> 9, n = gr & (NN - 1);
      xkT[((size_t)n * BATCH + b) * GATE + nt * 16 + col] = acc[v] + bias;
    }
  }
}

// ---------------------------------------------------------------- LSTM recurrence
// One workgroup, 8 waves; h A-swizzled in LDS; z = xk_t + h@rk via 32 WMMA/step.
__global__ void k_lstm(const float* __restrict__ xkT, const _Float16* __restrict__ lrks,
                       float* __restrict__ bb) {
  __shared__ __align__(32) _Float16 hs[4 * 512];   // 4 batch tiles, A-swizzled
  __shared__ float cst[BATCH * HD];
  __shared__ float zbuf[BATCH * GATE];
  const int tid  = threadIdx.x;
  const int wave = tid >> 5;
  const int mt   = wave & 3;               // batch-row tile
  const int ch   = wave >> 2;              // gate-column half
  for (int i = tid; i < 4 * 512; i += 256) hs[i] = (_Float16)0.0f;
  for (int i = tid; i < BATCH * HD; i += 256) cst[i] = 0.0f;
  __syncthreads();
  for (int t = 0; t < NN; ++t) {
    v16h a = load_frag(hs + mt * 512);
#pragma unroll
    for (int ct = 0; ct < 4; ++ct) {
      const int colt = ch * 64 + ct * 16;
      v8f acc = load_c_f32(xkT + ((size_t)t * BATCH + mt * 16) * GATE + colt, GATE);
      acc = wmma16(a, load_frag(lrks + (colt >> 4) * 512), acc);
      store_d_f32(zbuf + mt * 16 * GATE + colt, GATE, acc);
    }
    __syncthreads();
#pragma unroll
    for (int e = 0; e < 8; ++e) {
      const int cell = tid * 8 + e;        // 2048 cells
      const int bi = cell >> 5, j = cell & 31;
      float zi = zbuf[bi * GATE + j];
      float zf = zbuf[bi * GATE + j + HD];
      float zg = zbuf[bi * GATE + j + 2 * HD];
      float zo = zbuf[bi * GATE + j + 3 * HD];
      float cn = sigm(zf) * cst[bi * HD + j] + sigm(zi) * tanhf(zg);
      float hn = sigm(zo) * tanhf(cn);
      cst[bi * HD + j] = cn;
      // h stored A-swizzled: rows = batch, K = hidden j
      hs[(bi >> 4) * 512 + ((bi & 15) + (((j >> 3) & 1) << 4)) * 16
         + (j & 7) + ((j >> 4) << 3)] = (_Float16)hn;
      bb[((size_t)bi * NN + t) * ED + HD + j] = hn;
    }
    __syncthreads();
  }
}

// ---------------------------------------------------------------- t = adj @ x
// All operands pre-swizzled; every fragment = one 32B contiguous load per lane.
// Output written A-swizzled for the dense layers.
__global__ void k_adj_gemm(const _Float16* __restrict__ adjs,
                           const _Float16* __restrict__ xswz,
                           _Float16* __restrict__ tswz) {
  const int gw = blockIdx.x * 4 + (threadIdx.x >> 5);
  const int b  = gw >> 5;                  // 32 row tiles per batch
  const _Float16* abase = adjs + ((size_t)gw << 13);        // 16 tiles * 512
  const _Float16* xb    = xswz + ((size_t)b << 15);         // 64 tiles * 512
  v8f acc0 = {}, acc1 = {}, acc2 = {}, acc3 = {};
  for (int ks = 0; ks < NN / 32; ++ks) {
    __builtin_prefetch(abase + ((ks + 1) << 9), 0, 1);
    v16h a = load_frag(abase + (ks << 9));
    acc0 = wmma16(a, load_frag(xb + ((ks * 4 + 0) << 9)), acc0);
    acc1 = wmma16(a, load_frag(xb + ((ks * 4 + 1) << 9)), acc1);
    acc2 = wmma16(a, load_frag(xb + ((ks * 4 + 2) << 9)), acc2);
    acc3 = wmma16(a, load_frag(xb + ((ks * 4 + 3) << 9)), acc3);
  }
  _Float16* tb = tswz + (size_t)gw * 1024;  // 2 A-tiles per 16-row block
  store_d_aswz(tb, 0,  acc0);
  store_d_aswz(tb, 16, acc1);
  store_d_aswz(tb, 32, acc2);
  store_d_aswz(tb, 48, acc3);
}

// ---------------------------------------------------------------- dense x2 + tanh
// u = relu(t@w0+b0); v = u@w1+b1; x = tanh(bb + v) -> x32 (row-major) + xswz.
__global__ void k_dense_tanh(const _Float16* __restrict__ tswz,
                             const _Float16* __restrict__ w0s,
                             const _Float16* __restrict__ w1s,
                             const float* __restrict__ we_b,
                             const float* __restrict__ bb,
                             float* __restrict__ x32, _Float16* __restrict__ xswz) {
  __shared__ __align__(32) _Float16 u[4][1024];   // per-wave, A-swizzled
  const int wave = threadIdx.x >> 5;
  const int gw   = blockIdx.x * 4 + wave;
  const int R    = gw * 16;
  const int lane = threadIdx.x & 31;
  const int col  = lane & 15;
  const int rb   = (lane >> 4) << 3;
  v16h a0 = load_frag(tswz + (size_t)gw * 1024);
  v16h a1 = load_frag(tswz + (size_t)gw * 1024 + 512);
#pragma unroll
  for (int nt = 0; nt < 4; ++nt) {
    v8f acc = {};
    acc = wmma16(a0, load_frag(w0s + (0 * 4 + nt) * 512), acc);
    acc = wmma16(a1, load_frag(w0s + (1 * 4 + nt) * 512), acc);
    float bias = we_b[nt * 16 + col];
#pragma unroll
    for (int v = 0; v < 8; ++v) acc[v] = fmaxf(acc[v] + bias, 0.0f);
    store_d_aswz(u[wave], nt * 16, acc);
  }
  __syncthreads();
  v16h ua0 = load_frag(u[wave]), ua1 = load_frag(u[wave] + 512);
#pragma unroll
  for (int nt = 0; nt < 4; ++nt) {
    v8f acc = {};
    acc = wmma16(ua0, load_frag(w1s + (0 * 4 + nt) * 512), acc);
    acc = wmma16(ua1, load_frag(w1s + (1 * 4 + nt) * 512), acc);
    const int cg = nt * 16 + col;
    const float bias = we_b[ED + cg];
#pragma unroll
    for (int v = 0; v < 8; ++v) {
      const int gr = R + rb + v;
      const size_t idx = (size_t)gr * ED + cg;
      float val = tanhf(bb[idx] + acc[v] + bias);
      x32[idx] = val;
      const int b = gr >> 9, n = gr & (NN - 1);
      const int kb = n >> 5, kl = n & 31;
      xswz[(((size_t)(b * 16 + kb) * 4 + nt) << 9)
           + ((((kl >> 3) & 1) << 4) + col) * 16 + (kl & 7) + ((kl >> 4) << 3)] = (_Float16)val;
    }
  }
}

// ---------------------------------------------------------------- reduce + out proj
__global__ void k_reduce(const float* __restrict__ x32, const float* __restrict__ wout,
                         const float* __restrict__ bout, float* __restrict__ e) {
  __shared__ float sv[ED];
  const int b = blockIdx.x, d = threadIdx.x;
  float s = 0.0f;
  for (int n = 0; n < NN; ++n) s += x32[((size_t)b * NN + n) * ED + d];
  sv[d] = s;
  __syncthreads();
  float acc = bout[d];
  for (int k = 0; k < ED; ++k) acc += sv[k] * wout[k * OD + d];
  e[b * OD + d] = acc;
}

__global__ void k_cosine(const float* __restrict__ e1, const float* __restrict__ e2,
                         float* __restrict__ out) {
  const int b = threadIdx.x;
  float num = 0.0f, n1 = 0.0f, n2 = 0.0f;
  for (int d = 0; d < OD; ++d) {
    float a = e1[b * OD + d], c = e2[b * OD + d];
    num += a * c; n1 += a * a; n2 += c * c;
  }
  out[b] = num / sqrtf(n1 * n2 + 1e-10f);
}

// ---------------------------------------------------------------- host
extern "C" void kernel_launch(void* const* d_in, const int* in_sizes, int n_in,
                              void* d_out, int out_size, void* d_ws, size_t ws_size,
                              hipStream_t stream) {
  (void)in_sizes; (void)n_in; (void)out_size; (void)ws_size;
  const int*   cfg[2] = {(const int*)d_in[0],   (const int*)d_in[4]};
  const float* lit[2] = {(const float*)d_in[2], (const float*)d_in[6]};
  const float* sem[2] = {(const float*)d_in[3], (const float*)d_in[7]};
  const float* d1w[2] = {(const float*)d_in[8],  (const float*)d_in[15]};
  const float* d1b[2] = {(const float*)d_in[9],  (const float*)d_in[16]};
  const float* d2w[2] = {(const float*)d_in[10], (const float*)d_in[17]};
  const float* d2b[2] = {(const float*)d_in[11], (const float*)d_in[18]};
  const float* lk[2]  = {(const float*)d_in[12], (const float*)d_in[19]};
  const float* lrk[2] = {(const float*)d_in[13], (const float*)d_in[20]};
  const float* lb[2]  = {(const float*)d_in[14], (const float*)d_in[21]};
  const float* we_w   = (const float*)d_in[22];
  const float* we_b   = (const float*)d_in[23];
  const float* wout   = (const float*)d_in[24];
  const float* bout   = (const float*)d_in[25];

  char* p = (char*)d_ws;
  auto carve = [&](size_t bytes) -> char* {
    char* r = p; p += (bytes + 255) & ~(size_t)255; return r;
  };
  _Float16 *d1wS[2], *d2wS[2], *lkS[2], *lrkS[2];
  for (int br = 0; br < 2; ++br) {
    d1wS[br] = (_Float16*)carve((size_t)LITW * ED * 2);
    d2wS[br] = (_Float16*)carve((size_t)ED * HD * 2);
    lkS[br]  = (_Float16*)carve((size_t)SEMW * GATE * 2);
    lrkS[br] = (_Float16*)carve((size_t)HD * GATE * 2);
  }
  _Float16* weS  = (_Float16*)carve((size_t)2 * ED * ED * 2);
  _Float16* adjs = (_Float16*)carve((size_t)BATCH * NN * NN * 2);
  float*    bb   = (float*)carve((size_t)BATCH * NN * ED * 4);
  float*    xkT  = (float*)carve((size_t)BATCH * NN * GATE * 4);
  _Float16* tswz = (_Float16*)carve((size_t)BATCH * NN * ED * 2);
  float*    x32  = (float*)carve((size_t)BATCH * NN * ED * 4);
  _Float16* xswz = (_Float16*)carve((size_t)BATCH * NN * ED * 2);
  float* ebuf[2] = {(float*)carve((size_t)BATCH * OD * 4),
                    (float*)carve((size_t)BATCH * OD * 4)};

  auto swzB = [&](const float* s, _Float16* d, int K, int N) {
    int n = K * N;
    k_swz_b<<<(n + 255) / 256, 256, 0, stream>>>(s, d, K, N);
  };
  for (int br = 0; br < 2; ++br) {
    swzB(d1w[br], d1wS[br], LITW, ED);
    swzB(d2w[br], d2wS[br], ED, HD);
    swzB(lk[br],  lkS[br],  SEMW, GATE);
    swzB(lrk[br], lrkS[br], HD, GATE);
  }
  swzB(we_w,            weS,            ED, ED);   // layer 0
  swzB(we_w + ED * ED,  weS + ED * ED,  ED, ED);   // layer 1

  const int total = BATCH * NN * ED;
  for (int br = 0; br < 2; ++br) {
    k_swz_adj<<<(int)(((size_t)BATCH * NN * NN) / 256), 256, 0, stream>>>(cfg[br], adjs);
    k_mlp<<<BATCH * NN / 16, 32, 0, stream>>>(lit[br], d1wS[br], d1b[br],
                                              d2wS[br], d2b[br], bb);
    k_xk<<<BATCH * NN / 16, 32, 0, stream>>>(sem[br], lkS[br], lb[br], xkT);
    k_lstm<<<1, 256, 0, stream>>>(xkT, lrkS[br], bb);
    k_init_x<<<total / 256, 256, 0, stream>>>(bb, xswz);
    for (int it = 0; it < ITERS; ++it) {
      k_adj_gemm<<<BATCH * (NN / 16) / 4, 128, 0, stream>>>(adjs, xswz, tswz);
      k_dense_tanh<<<(BATCH * NN / 16) / 4, 128, 0, stream>>>(
          tswz, weS, weS + ED * ED, we_b, bb, x32, xswz);
    }
    k_reduce<<<BATCH, ED, 0, stream>>>(x32, wout, bout, ebuf[br]);
  }
  k_cosine<<<1, BATCH, 0, stream>>>(ebuf[0], ebuf[1], (float*)d_out);
}
